// GeneralAttention_3796751089825
// MI455X (gfx1250) — matrix operand order repaired and checked by reading, not compiled, not run on hardware
//
#include <hip/hip_runtime.h>

// ---------------------------------------------------------------------------
// GeneralAttention for MI455X (gfx1250), wave32 + WMMA f16 16x16x32 + TDM.
//   Q = X @ W^T + b            (proj kernel, WMMA)
//   S = exp(Q @ X^T) * causal  (attn kernel, WMMA + v_exp_f32)
//   out = (S / rowsum(S)) @ X  (attn kernel, WMMA; rowsum via WMMA with B=1s)
// Key tiles are staged global->LDS by the Tensor Data Mover (double-buffered,
// overlapped with WMMA compute), falling back to register staging when the
// TDM builtin is unavailable (e.g. host pass).
// ---------------------------------------------------------------------------

typedef __attribute__((ext_vector_type(16))) _Float16 v16h;
typedef __attribute__((ext_vector_type(8)))  _Float16 v8h;
typedef __attribute__((ext_vector_type(4)))  _Float16 v4h;
typedef __attribute__((ext_vector_type(8)))  float    v8f;
typedef __attribute__((ext_vector_type(4)))  float    v4f;

#define B_DIM 8
#define S_DIM 2048
#define H_DIM 512

#if defined(__AMDGCN__)
#if __has_builtin(__builtin_amdgcn_tensor_load_to_lds)
#define USE_TDM 1
#endif
#endif
#ifndef USE_TDM
#define USE_TDM 0
#endif

#if USE_TDM
typedef __attribute__((ext_vector_type(4))) unsigned int u32x4;
typedef __attribute__((ext_vector_type(8))) int          i32x8;
typedef __attribute__((ext_vector_type(4))) int          i32x4;

// Build a 2D f16 tile descriptor (D#) per CDNA5 ISA ch.8 and issue the DMA.
//   group0: count=1 | lds_addr | global_addr[56:0] | type=2
//   group1: data_size=2B | tensor_dim0/1 | tile_dim0/1 | tensor_dim0_stride
static __device__ inline void tdm_load_tile_f16(const _Float16* gsrc,
                                                const _Float16* ldst,
                                                int rows, int cols,
                                                int rowStrideElems) {
  const unsigned long long ga = (unsigned long long)(size_t)gsrc;
  u32x4 g0;
  g0.x = 1u;                                   // count=1, user load descriptor
  g0.y = (unsigned)(size_t)(const void*)ldst;  // LDS byte address
  g0.z = (unsigned)ga;                         // global_addr[31:0]
  g0.w = (unsigned)((ga >> 32) & 0x01FFFFFFu) | (2u << 30);  // [56:32]|type=2
  i32x8 g1;
  g1[0] = 1 << 16;                                             // data_size=2B
  g1[1] = (int)(((unsigned)cols & 0xFFFFu) << 16);             // t_dim0 lo16
  g1[2] = (int)(((unsigned)cols >> 16) |
                (((unsigned)rows & 0xFFFFu) << 16));           // d0 hi | d1 lo
  g1[3] = (int)(((unsigned)rows >> 16) | ((unsigned)cols << 16)); // | tile_d0
  g1[4] = rows & 0xFFFF;                                       // tile_dim1
  g1[5] = rowStrideElems;                                      // d0_stride lo
  g1[6] = 0;
  g1[7] = 0;
  i32x4 zz = {0, 0, 0, 0};
#if __clang_major__ >= 23
  i32x8 z8 = {0, 0, 0, 0, 0, 0, 0, 0};
  __builtin_amdgcn_tensor_load_to_lds(g0, g1, zz, zz, z8, 0);
#else
  __builtin_amdgcn_tensor_load_to_lds(g0, g1, zz, zz, 0);
#endif
}

#if __has_builtin(__builtin_amdgcn_s_wait_tensorcnt)
#define TDM_WAIT(n) __builtin_amdgcn_s_wait_tensorcnt(n)
#else
#define TDM_WAIT(n) asm volatile("s_wait_tensorcnt " #n ::: "memory")
#endif
#endif  // USE_TDM

// ---- WMMA fragment loaders (wave32 layouts per CDNA5 ISA 7.12.2) -----------
// A (16xK=32, f16): lane L -> row m = L&15; half = L>>4 selects K-phase.
static __device__ inline v16h load_a_frag(const _Float16* base, int ld,
                                          int row0, int k0, int lane) {
  const int m  = lane & 15;
  const int hh = lane >> 4;
  const _Float16* p = base + (size_t)(row0 + m) * ld + k0 + 8 * hh;
  v8h lo = *(const v8h*)(p);
  v8h hi = *(const v8h*)(p + 16);
  v16h r;
#pragma unroll
  for (int i = 0; i < 8; ++i) { r[i] = lo[i]; r[8 + i] = hi[i]; }
  return r;
}

// B (K=32 x 16, f16) from a row-major buffer whose ROWS are B's columns
// (i.e. B^T row-major): lane n reads halves at k = 8*(L>>4) and 16 + 8*(L>>4).
static __device__ inline v16h load_b_frag(const _Float16* baseT, int ld,
                                          int n0, int k0, int lane) {
  const int n  = lane & 15;
  const int kq = lane >> 4;
  const _Float16* p = baseT + (size_t)(n0 + n) * ld + k0 + 8 * kq;
  v8h lo = *(const v8h*)(p);
  v8h hi = *(const v8h*)(p + 16);
  v16h r;
#pragma unroll
  for (int i = 0; i < 8; ++i) { r[i] = lo[i]; r[8 + i] = hi[i]; }
  return r;
}

static __device__ inline v8f wmma_f16(v16h a, v16h b, v8f c) {
  return __builtin_amdgcn_wmma_f32_16x16x32_f16(
      false, a, false, b, (short)0, c, false, false);
}

// ---- stage 1: f32 -> f16 conversion ---------------------------------------
__global__ __launch_bounds__(256) void cvt_kernel(const float* __restrict__ src,
                                                  _Float16* __restrict__ dst,
                                                  int n4) {
  int i = blockIdx.x * blockDim.x + threadIdx.x;
  const int stride = gridDim.x * blockDim.x;
  for (; i < n4; i += stride) {
    v4f v = ((const v4f*)src)[i];
    v4h o;
#pragma unroll
    for (int j = 0; j < 4; ++j) o[j] = (_Float16)v[j];
    ((v4h*)dst)[i] = o;
  }
}

// ---- stage 2: Q = X @ W^T + b  (M=B*S=16384, N=K=H=512) -------------------
__global__ __launch_bounds__(256) void proj_kernel(
    const _Float16* __restrict__ Xh, const _Float16* __restrict__ Wh,
    const float* __restrict__ bias, _Float16* __restrict__ Qh) {
  const int lane = threadIdx.x & 31;
  const int wave = threadIdx.x >> 5;
  const int tile = blockIdx.x * 8 + wave;      // 8 waves per block
  const int NT   = H_DIM / 16;                 // 32 n-tiles
  const int m0   = (tile / NT) * 16;
  const int n0   = (tile % NT) * 16;

  v8f acc = {};
#pragma unroll
  for (int kb = 0; kb < H_DIM / 32; ++kb) {
    v16h a = load_a_frag(Xh, H_DIM, m0, kb * 32, lane);
    v16h b = load_b_frag(Wh, H_DIM, n0, kb * 32, lane);
    acc = wmma_f16(a, b, acc);
  }
  const int n  = lane & 15;
  const int hh = lane >> 4;
  const float bv = bias[n0 + n];
#pragma unroll
  for (int v = 0; v < 8; ++v)
    Qh[(size_t)(m0 + v + 8 * hh) * H_DIM + n0 + n] = (_Float16)(acc[v] + bv);
}

// ---- stage 3: flash-attention over key blocks of 32 -----------------------
// 256 threads = 8 waves. Workgroup owns 32 query rows (2 row-tiles of 16).
// wave = (r<<2)|c : r = query row-tile (0..1), c = output h-quarter (0..3).
__global__ __launch_bounds__(256) void attn_kernel(
    const _Float16* __restrict__ Xh, const _Float16* __restrict__ Qh,
    float* __restrict__ out) {
  extern __shared__ _Float16 lds[];
  _Float16* Qs = lds;                  // 32 x 512   staged Q block
  _Float16* Xk = lds + 32 * H_DIM;     // 2x 32x512  key block (double-buffer)
  _Float16* Xt = lds + 96 * H_DIM;     // 512 x 32   key block transposed (V)
  _Float16* Ps = lds + 128 * H_DIM;    // 8 x (16x32) per-wave exp(S) scratch

  const int tid  = threadIdx.x;
  const int lane = tid & 31;
  const int wave = tid >> 5;
  const int r    = wave >> 2;
  const int c    = wave & 3;
  const int qb   = blockIdx.x & (S_DIM / 32 - 1);
  const int bidx = blockIdx.x >> 6;
  const int q0   = qb * 32;
  const _Float16* Xb = Xh + (size_t)bidx * S_DIM * H_DIM;
  const _Float16* Qg = Qh + ((size_t)bidx * S_DIM + q0) * H_DIM;

#if USE_TDM
  // TDM prologue: DMA Q block + key block 0 (in-order on TENSORcnt).
  if (wave == 0) {
    tdm_load_tile_f16(Qg, Qs, 32, H_DIM, H_DIM);
    tdm_load_tile_f16(Xb, Xk, 32, H_DIM, H_DIM);
  }
#else
  for (int ci = tid; ci < 32 * H_DIM / 8; ci += 256)
    ((v8h*)Qs)[ci] = ((const v8h*)Qg)[ci];
#endif

  v8f oacc[8] = {};
  v8f denom   = {};
  v16h ones;
#pragma unroll
  for (int i = 0; i < 16; ++i) ones[i] = (_Float16)1.0f;

  const int n     = lane & 15;
  const int hh    = lane >> 4;
  const int iBase = q0 + 16 * r + 8 * hh;   // query row for accum slot v

  for (int jb = 0; jb <= qb; ++jb) {
    const int j0 = jb * 32;
#if USE_TDM
    _Float16* Xkc = Xk + (jb & 1) * (32 * H_DIM);
    __syncthreads();  // all waves done reading both LDS key buffers
    if (wave == 0) {
      if (jb < qb)    // overlap next tile's DMA with this tile's WMMAs
        tdm_load_tile_f16(Xb + (size_t)(j0 + 32) * H_DIM,
                          Xk + ((jb + 1) & 1) * (32 * H_DIM), 32, H_DIM,
                          H_DIM);
      if (jb < qb) TDM_WAIT(1);  // tile jb (and Q on iter 0) complete
      else         TDM_WAIT(0);
    }
    __syncthreads();  // tile jb visible to all waves
    // build transposed copy (for the P@V B-operand) from LDS
    for (int ci = tid; ci < 32 * H_DIM / 8; ci += 256) {
      const int row = ci >> 6;
      const int col = (ci & 63) * 8;
      v8h v = *(const v8h*)(Xkc + row * H_DIM + col);
#pragma unroll
      for (int e = 0; e < 8; ++e) Xt[(col + e) * 32 + row] = v[e];
    }
    __syncthreads();
#else
    _Float16* Xkc = Xk;
    __syncthreads();
    const _Float16* Kg = Xb + (size_t)j0 * H_DIM;
    for (int ci = tid; ci < 32 * H_DIM / 8; ci += 256) {
      const int row = ci >> 6;
      const int col = (ci & 63) * 8;
      v8h v = ((const v8h*)Kg)[ci];
      *(v8h*)(Xkc + row * H_DIM + col) = v;
#pragma unroll
      for (int e = 0; e < 8; ++e) Xt[(col + e) * 32 + row] = v[e];
    }
    if (jb < qb)
      __builtin_prefetch(Xb + (size_t)(j0 + 32) * H_DIM + tid * 64, 0, 0);
    __syncthreads();
#endif

    // scores: 16x32 tile of Q . K^T, full K=512 contraction
    v8f s0 = {}, s1 = {};
#pragma unroll
    for (int kb = 0; kb < H_DIM / 32; ++kb) {
      v16h a  = load_a_frag(Qs, H_DIM, 16 * r, kb * 32, lane);
      v16h b0 = load_b_frag(Xkc, H_DIM, 0,  kb * 32, lane);
      v16h b1 = load_b_frag(Xkc, H_DIM, 16, kb * 32, lane);
      s0 = wmma_f16(a, b0, s0);
      s1 = wmma_f16(a, b1, s1);
    }

    // strictly-causal mask (j < i) + exp; spill P=exp(S) to LDS in A layout
    _Float16* P = Ps + wave * (16 * 32);
#pragma unroll
    for (int v = 0; v < 8; ++v) {
      const int i = iBase + v;
      const float e0 = (j0 + n      < i) ? __expf(s0[v]) : 0.0f;
      const float e1 = (j0 + n + 16 < i) ? __expf(s1[v]) : 0.0f;
      P[(v + 8 * hh) * 32 + n]      = (_Float16)e0;
      P[(v + 8 * hh) * 32 + n + 16] = (_Float16)e1;
    }
    // same-wave LDS RAW below: DScnt wait inserted by compiler

    v16h pa = load_a_frag(P, 32, 0, 0, lane);
    denom = wmma_f16(pa, ones, denom);   // row-sums, lane-aligned with oacc
#pragma unroll
    for (int t = 0; t < 8; ++t) {
      v16h bv = load_b_frag(Xt, 32, 128 * c + 16 * t, 0, lane);
      oacc[t] = wmma_f16(pa, bv, oacc[t]);
    }
  }

  // normalize + store (row 0 => denom==0 => out==0, matches reference eps)
  float* Ob = out + (size_t)bidx * S_DIM * H_DIM;
#pragma unroll
  for (int v = 0; v < 8; ++v) {
    const int row = iBase + v;
    const float inv = 1.0f / (denom[v] + 1e-10f);
#pragma unroll
    for (int t = 0; t < 8; ++t)
      Ob[(size_t)row * H_DIM + 128 * c + 16 * t + n] = oacc[t][v] * inv;
  }
}

// ---------------------------------------------------------------------------
extern "C" void kernel_launch(void* const* d_in, const int* in_sizes, int n_in,
                              void* d_out, int out_size, void* d_ws,
                              size_t ws_size, hipStream_t stream) {
  (void)in_sizes; (void)n_in; (void)out_size; (void)ws_size;
  const float* X    = (const float*)d_in[0];
  const float* W    = (const float*)d_in[1];
  const float* bias = (const float*)d_in[2];
  float* out        = (float*)d_out;

  _Float16* Xh = (_Float16*)d_ws;                        // 16 MB
  _Float16* Wh = Xh + (size_t)B_DIM * S_DIM * H_DIM;     // 0.5 MB
  _Float16* Qh = Wh + (size_t)H_DIM * H_DIM;             // 16 MB

  const int nX4 = B_DIM * S_DIM * H_DIM / 4;
  const int nW4 = H_DIM * H_DIM / 4;
  cvt_kernel<<<1024, 256, 0, stream>>>(X, Xh, nX4);
  cvt_kernel<<<64,   256, 0, stream>>>(W, Wh, nW4);

  const int tiles = (B_DIM * S_DIM / 16) * (H_DIM / 16); // 32768
  proj_kernel<<<tiles / 8, 256, 0, stream>>>(Xh, Wh, bias, Qh);

  // LDS: Qs(32K) + Xk double buffer(64K) + Xt(32K) + Ps(8K) = 136 KB
  const int ldsBytes = (128 * H_DIM + 8 * 16 * 32) * (int)sizeof(_Float16);
  attn_kernel<<<B_DIM * (S_DIM / 32), 256, ldsBytes, stream>>>(Xh, Qh, out);
}
